// RUMambaTimeSeries_30064771072273
// MI455X (gfx1250) — compile-verified, matrix operand order checked by reading, not run on hardware
//
#include <hip/hip_runtime.h>
#include <hip/hip_bf16.h>

// ---------------------------------------------------------------------------
// RUMambaTimeSeries — MI455X (gfx1250, wave32, WMMA + async-LDS) implementation
//
// Shapes: B=64 T=16 DM=1024 D_INNER=2048 NHEADS=32 HEADDIM=64 D_STATE=128
//         CONV_DIM=2304 D_IN_PROJ=4384 NUM_CLASS=2513
//
// Algebraic collapses:
//  * X[i,b,j] = (j>=i) ? inputs[b,i] : 0  -> in-proj GEMM is 1024x4384x1024.
//  * Only hidden[:, -1, :] is used        -> SSM reduces to per-seq vector ops.
//
// GEMM data path: double-buffered LDS tiles filled with
// global_load_async_to_lds_b128 (ASYNCcnt), s_wait_asynccnt + barrier, then
// v_wmma_f32_16x16x32_f16 on fragments read from LDS.
// ---------------------------------------------------------------------------

typedef __attribute__((ext_vector_type(16))) _Float16 v16h;
typedef __attribute__((ext_vector_type(8)))  float    v8f;

#define BSZ      64
#define TSEQ     16
#define DMODEL   1024
#define DSTATE   128
#define HEADDIM  64
#define DINNER   2048
#define NHEADS   32
#define CONVDIM  2304
#define DINPROJ  4384
#define NCLS     2513
#define NSEQ     1024     // T * B

// -------------------------------- helpers ----------------------------------

__device__ __forceinline__ float siluf(float x) {
  return x * (1.0f / (1.0f + __expf(-x)));
}
__device__ __forceinline__ float softplusf(float x) {
  return (x > 20.0f) ? x : log1pf(__expf(x));
}

// 16B-per-lane async copy global -> LDS (ASYNCcnt).  vdst = LDS byte address
// (low 32 bits of a generic pointer to a __shared__ object), vaddr = 64-bit
// global address, no saddr.
__device__ __forceinline__ void async_g2l_b128(unsigned lds_off, const void* g) {
  asm volatile("global_load_async_to_lds_b128 %0, %1, off"
               :: "v"(lds_off), "v"(g) : "memory");
}
__device__ __forceinline__ void wait_async0() {
  asm volatile("s_wait_asynccnt 0x0" ::: "memory");
}
__device__ __forceinline__ unsigned lds_off_of(const void* p) {
  return (unsigned)(unsigned long long)p;   // low 32 bits = LDS byte offset
}

// 16-bit 16x32 WMMA fragment gather (row-major source in LDS, ld=32 halfs),
// per ISA 7.12.2: lane&15 = row, lane>>4 selects K-half (+8). VGPR pair v ->
// K base {0,2,4,6,16,18,20,22}.  Same layout serves B^T frags for NT GEMM.
__device__ __forceinline__ v16h frag16_lds(const _Float16* p /*row base + khalf*/) {
  v16h f;
#pragma unroll
  for (int v = 0; v < 8; ++v) {
    int kb = (v < 4) ? (2 * v) : (16 + 2 * (v - 4));
    f[2 * v]     = p[kb];
    f[2 * v + 1] = p[kb + 1];
  }
  return f;
}

// --------------------------- f32 -> f16 convert ----------------------------

__global__ void cvt_f32_to_f16(const float* __restrict__ s,
                               _Float16* __restrict__ d, int n) {
  int i = blockIdx.x * blockDim.x + threadIdx.x;
  int stride = gridDim.x * blockDim.x;
  for (; i < n; i += stride) d[i] = (_Float16)s[i];
}

// ------------------------------ WMMA NT GEMM -------------------------------
// C[M,N] = A[M,K] * Bm[N,K]^T (+ bias).  Row-major everywhere.
// Block = 256 threads = 8 waves: 2 (M) x 4 (N); wave tile 16x64.
// Block tile 32 x 256, K step 32, double-buffered async LDS staging.
// permute!=0 remaps output row n=i*64+b -> b*16+i.

__global__ void __launch_bounds__(256)
wmma_gemm_nt(const _Float16* __restrict__ A, const _Float16* __restrict__ Bm,
             float* __restrict__ C, const float* __restrict__ bias,
             int M, int N, int K, int lda, int ldb, int ldc, int permute) {
  __shared__ _Float16 sA[2][32][32];    //  2 KB x2
  __shared__ _Float16 sB[2][256][32];   // 16 KB x2

  const int tid   = threadIdx.x;
  const int lane  = tid & 31;
  const int wave  = tid >> 5;
  const int wm    = wave >> 2;          // 0..1
  const int wn    = wave & 3;           // 0..3
  const int mblk  = blockIdx.x * 32;
  const int nblk  = blockIdx.y * 256;

  // staging indices: A tile = 32 rows x 64B (128 x 16B chunks, threads 0..127)
  const int aro = tid >> 2;             // 0..63 (only <32 used when tid<128)
  const int ach = tid & 3;
  int agr = mblk + aro; if (agr >= M) agr = M - 1;
  // B tile = 256 rows x 64B (1024 x 16B chunks, 4 per thread)
  int bro[4], bch[4], bgr[4];
#pragma unroll
  for (int q = 0; q < 4; ++q) {
    int lin = tid + 256 * q;
    bro[q] = lin >> 2;
    bch[q] = lin & 3;
    int g = nblk + bro[q];
    bgr[q] = (g < N) ? g : (N - 1);
  }

  auto stage = [&](int k, int buf) {
    if (tid < 128)
      async_g2l_b128(lds_off_of(&sA[buf][aro][ach * 8]),
                     A + (size_t)agr * lda + k + ach * 8);
#pragma unroll
    for (int q = 0; q < 4; ++q)
      async_g2l_b128(lds_off_of(&sB[buf][bro[q]][bch[q] * 8]),
                     Bm + (size_t)bgr[q] * ldb + k + bch[q] * 8);
  };

  const int frow  = lane & 15;
  const int khalf = (lane >> 4) * 8;

  v8f acc[4];
#pragma unroll
  for (int j = 0; j < 4; ++j)
#pragma unroll
    for (int r = 0; r < 8; ++r) acc[j][r] = 0.0f;

  stage(0, 0);
  wait_async0();
  __syncthreads();

  int buf = 0;
  for (int k = 0; k < K; k += 32) {
    if (k + 32 < K) stage(k + 32, buf ^ 1);   // overlap fill with math

    v16h af = frag16_lds(&sA[buf][wm * 16 + frow][khalf]);
#pragma unroll
    for (int j = 0; j < 4; ++j) {
      v16h bf = frag16_lds(&sB[buf][wn * 64 + j * 16 + frow][khalf]);
      acc[j] = __builtin_amdgcn_wmma_f32_16x16x32_f16(
          /*neg_a=*/false, af, /*neg_b=*/false, bf,
          /*c_mod=*/(short)0, acc[j], /*reuse_a=*/false, /*reuse_b=*/false);
    }

    wait_async0();
    __syncthreads();
    buf ^= 1;
  }

  // D layout: lane&15 = N col, lane>>4 -> rows +8, VGPR r -> row r
  const int col  = lane & 15;
  const int rowb = (lane >> 4) * 8;
#pragma unroll
  for (int j = 0; j < 4; ++j) {
    int nn = nblk + wn * 64 + j * 16 + col;
    if (nn >= N) continue;
    float bv = bias ? bias[nn] : 0.0f;
#pragma unroll
    for (int r = 0; r < 8; ++r) {
      int rr = mblk + wm * 16 + rowb + r;
      if (rr >= M) continue;
      int orow = permute ? ((rr & 63) * TSEQ + (rr >> 6)) : rr;
      C[(size_t)orow * ldc + nn] = acc[j][r] + bv;
    }
  }
}

// ------------------- fused conv/SSM/gate/RMSnorm kernel --------------------
// One 256-thread block per sequence n = i*64 + b.  All timestep values derive
// from the single projected vector v = proj[b*16+i, :] (4384 floats).

__global__ void __launch_bounds__(256)
mamba_mid(const float* __restrict__ proj, const float* __restrict__ conv_w,
          const float* __restrict__ conv_b, const float* __restrict__ dt_bias,
          const float* __restrict__ A_log, const float* __restrict__ D_skip,
          const float* __restrict__ norm_w, _Float16* __restrict__ hidden) {
  const int n   = blockIdx.x;       // sequence id = i*64 + b
  const int i   = n >> 6;
  const int b   = n & 63;
  const int tid = threadIdx.x;
  const float* __restrict__ v = proj + (size_t)(b * TSEQ + i) * DINPROJ;

  __shared__ float s_sp[NHEADS], s_gh[NHEADS];
  __shared__ float s_Bm[TSEQ][DSTATE];
  __shared__ float s_C15[DSTATE];
  __shared__ float s_CB[TSEQ];
  __shared__ float s_w[TSEQ][NHEADS];
  __shared__ float s_red[8];
  __shared__ float s_rms;

  // per-head scalars
  if (tid < NHEADS) {
    float sp = softplusf(v[4352 + tid] + dt_bias[tid]);
    s_sp[tid] = sp;                       // dt value on valid timesteps
    s_gh[tid] = -__expf(A_log[tid]) * sp; // dt * A  (negative)
  }

  // B-state rows for all 16 timesteps: cc in [2048,2176)
  for (int idx = tid; idx < TSEQ * DSTATE; idx += 256) {
    int s = idx >> 7, d = idx & 127;
    int cc = 2048 + d;
    float vv = v[2048 + cc];
    float w0 = conv_w[cc * 4 + 0], w1 = conv_w[cc * 4 + 1];
    float w2 = conv_w[cc * 4 + 2], w3 = conv_w[cc * 4 + 3];
    int km = i - s + 3;
    float cs = (km <= 0) ? (w0 + w1 + w2 + w3)
             : (km == 1) ? (w1 + w2 + w3)
             : (km == 2) ? (w2 + w3)
             : (km == 3) ? w3 : 0.0f;
    s_Bm[s][d] = siluf(conv_b[cc] + vv * cs);
  }

  // C-state row at t = 15: cc in [2176,2304)
  if (tid < DSTATE) {
    int cc = 2176 + tid;
    float vv = v[2048 + cc];
    float w0 = conv_w[cc * 4 + 0], w1 = conv_w[cc * 4 + 1];
    float w2 = conv_w[cc * 4 + 2], w3 = conv_w[cc * 4 + 3];
    int km = i - 12;                       // km(15), in [-12,3]
    float cs = (km <= 0) ? (w0 + w1 + w2 + w3)
             : (km == 1) ? (w1 + w2 + w3)
             : (km == 2) ? (w2 + w3) : w3;
    s_C15[tid] = siluf(conv_b[cc] + vv * cs);
  }
  __syncthreads();

  // CB[s] = C15 . Bm[s]  (wave32 shuffle reduction, 2 rows per wave)
  {
    const int lane = tid & 31, wv = tid >> 5;
    for (int s = wv; s < TSEQ; s += 8) {
      float p = 0.0f;
#pragma unroll
      for (int q = 0; q < 4; ++q)
        p += s_C15[lane + 32 * q] * s_Bm[s][lane + 32 * q];
      p += __shfl_xor(p, 16);
      p += __shfl_xor(p, 8);
      p += __shfl_xor(p, 4);
      p += __shfl_xor(p, 2);
      p += __shfl_xor(p, 1);
      if (lane == 0) s_CB[s] = p;
    }
  }
  __syncthreads();

  // mixing weights w[s,h]
  for (int idx = tid; idx < TSEQ * NHEADS; idx += 256) {
    int s = idx >> 5, h = idx & 31;
    s_w[s][h] = (s >= i)
        ? s_CB[s] * __expf((float)(15 - s) * s_gh[h]) * s_sp[h]
        : 0.0f;
  }
  __syncthreads();

  // y15 -> gate -> accumulate sum of squares (8 channels per thread)
  float gv[8];
  float ss = 0.0f;
#pragma unroll
  for (int u = 0; u < 8; ++u) {
    int c = tid + 256 * u;                 // c in [0,2048)
    int h = c >> 6;
    float vv = v[2048 + c];
    float cb = conv_b[c];
    float w0 = conv_w[c * 4 + 0], w1 = conv_w[c * 4 + 1];
    float w2 = conv_w[c * 4 + 2], w3 = conv_w[c * 4 + 3];
    float cs0 = w0 + w1 + w2 + w3, cs1 = w1 + w2 + w3, cs2 = w2 + w3, cs3 = w3;
    float acc = 0.0f, xl = 0.0f;
#pragma unroll
    for (int s = 0; s < TSEQ; ++s) {
      int km = i - s + 3;
      float cs = (km <= 0) ? cs0 : (km == 1) ? cs1
               : (km == 2) ? cs2 : (km == 3) ? cs3 : 0.0f;
      float xs = siluf(cb + vv * cs);
      acc += s_w[s][h] * xs;
      if (s == 15) xl = xs;
    }
    float y = acc + xl * D_skip[h];
    float g = y * siluf(v[c]);             // gate with z (t=15 -> z = v)
    gv[u] = g;
    ss += g * g;
  }

  // block-wide RMS reduction
  {
    const int lane = tid & 31, wv = tid >> 5;
    ss += __shfl_xor(ss, 16);
    ss += __shfl_xor(ss, 8);
    ss += __shfl_xor(ss, 4);
    ss += __shfl_xor(ss, 2);
    ss += __shfl_xor(ss, 1);
    if (lane == 0) s_red[wv] = ss;
  }
  __syncthreads();
  if (tid == 0) {
    float t = 0.0f;
#pragma unroll
    for (int w = 0; w < 8; ++w) t += s_red[w];
    s_rms = rsqrtf(t * (1.0f / (float)DINNER) + 1e-5f);
  }
  __syncthreads();

  const float rms = s_rms;
  _Float16* __restrict__ hp = hidden + (size_t)n * DINNER;
#pragma unroll
  for (int u = 0; u < 8; ++u) {
    int c = tid + 256 * u;
    hp[c] = (_Float16)(gv[u] * rms * norm_w[c]);
  }
}

// ------------------------------ host launcher ------------------------------

extern "C" void kernel_launch(void* const* d_in, const int* in_sizes, int n_in,
                              void* d_out, int out_size, void* d_ws, size_t ws_size,
                              hipStream_t stream) {
  (void)in_sizes; (void)n_in; (void)out_size; (void)ws_size;

  const float* inputs  = (const float*)d_in[0]; // [64][16][1024]
  const float* W_in    = (const float*)d_in[1]; // [4384][1024]
  const float* conv_w  = (const float*)d_in[2]; // [2304][4]
  const float* conv_b  = (const float*)d_in[3]; // [2304]
  const float* dt_bias = (const float*)d_in[4]; // [32]
  const float* A_log   = (const float*)d_in[5]; // [32]
  const float* D_skip  = (const float*)d_in[6]; // [32]
  const float* norm_w  = (const float*)d_in[7]; // [2048]
  const float* cls_w   = (const float*)d_in[8]; // [2513][2048]
  const float* cls_b   = (const float*)d_in[9]; // [2513]
  float* out = (float*)d_out;                   // [64][16][2513]

  // workspace carve-up (256B aligned)
  char* ws = (char*)d_ws;
  size_t off = 0;
  auto carve = [&](size_t bytes) -> char* {
    char* p = ws + off;
    off = (off + bytes + 255) & ~(size_t)255;
    return p;
  };
  const int N_INP  = BSZ * TSEQ * DMODEL;  // 1,048,576
  const int N_WIN  = DINPROJ * DMODEL;     // 4,489,216
  const int N_CLS  = NCLS * DINNER;        // 5,146,624
  const int N_PROJ = NSEQ * DINPROJ;       // 4,489,216
  const int N_HID  = NSEQ * DINNER;        // 2,097,152

  _Float16* inp_h = (_Float16*)carve((size_t)N_INP * 2);
  _Float16* win_h = (_Float16*)carve((size_t)N_WIN * 2);
  _Float16* cls_h = (_Float16*)carve((size_t)N_CLS * 2);
  float*    projf = (float*)   carve((size_t)N_PROJ * 4);
  _Float16* hid_h = (_Float16*)carve((size_t)N_HID * 2);

  // 1) f32 -> f16 operand conversions
  cvt_f32_to_f16<<<(N_INP + 1023) / 1024, 256, 0, stream>>>(inputs, inp_h, N_INP);
  cvt_f32_to_f16<<<(N_WIN + 1023) / 1024, 256, 0, stream>>>(W_in,   win_h, N_WIN);
  cvt_f32_to_f16<<<(N_CLS + 1023) / 1024, 256, 0, stream>>>(cls_w,  cls_h, N_CLS);

  // 2) in-projection GEMM: proj[1024,4384] = inputs[1024,1024] @ W_in^T
  {
    dim3 grid(NSEQ / 32, (DINPROJ + 255) / 256);
    wmma_gemm_nt<<<grid, 256, 0, stream>>>(inp_h, win_h, projf, nullptr,
                                           NSEQ, DINPROJ, DMODEL,
                                           DMODEL, DMODEL, DINPROJ, 0);
  }

  // 3) fused conv / SSM / gate / RMSnorm (one block per sequence)
  mamba_mid<<<NSEQ, 256, 0, stream>>>(projf, conv_w, conv_b, dt_bias,
                                      A_log, D_skip, norm_w, hid_h);

  // 4) classifier GEMM with fused bias + (i,b)->(b,i) row permute into d_out
  {
    dim3 grid(NSEQ / 32, (NCLS + 255) / 256);
    wmma_gemm_nt<<<grid, 256, 0, stream>>>(hid_h, cls_h, out, cls_b,
                                           NSEQ, NCLS, DINNER,
                                           DINNER, DINNER, NCLS, 1);
  }
}
